// CILRSVAEModel_74466142978774
// MI455X (gfx1250) — compile-verified, matrix operand order checked by reading, not run on hardware
//
#include <hip/hip_runtime.h>

#define B_ROWS 65536
#define E_DIM  512
#define SL_DIM 128
#define H_DIM  256
#define NBR    6
#define D_DIM  640

// Async direct global->LDS staging if the toolchain exposes it (gfx1250).
#if defined(__has_builtin)
#if __has_builtin(__builtin_amdgcn_global_load_async_to_lds_b128) && \
    __has_builtin(__builtin_amdgcn_s_wait_asynccnt)
#define USE_ASYNC_LDS 1
#endif
#endif
#ifndef USE_ASYNC_LDS
#define USE_ASYNC_LDS 0
#endif

typedef __attribute__((ext_vector_type(16))) __bf16       v16bf;
typedef __attribute__((ext_vector_type(8)))  float        v8f;
typedef __attribute__((ext_vector_type(8)))  unsigned int v8u;
typedef __attribute__((ext_vector_type(4)))  int          v4i;
#if USE_ASYNC_LDS
typedef __attribute__((address_space(1))) v4i* gv4i_p;   // global int4*
typedef __attribute__((address_space(3))) v4i* lv4i_p;   // LDS int4*
#endif

// Native bf16 conversion: lowers to v_cvt_pk_bf16_f32 on gfx1250 (RNE).
__device__ __forceinline__ unsigned short bf16_bits(float x) {
    __bf16 h = (__bf16)x;
    return __builtin_bit_cast(unsigned short, h);
}
__device__ __forceinline__ unsigned int pk2(float lo, float hi) {
    return (unsigned int)bf16_bits(lo) | ((unsigned int)bf16_bits(hi) << 16);
}

// ---------------------------------------------------------------------------
// Prep A: dst[b][n][k] = bf16(src[b][k][n])   (transpose + f32->bf16)
// ---------------------------------------------------------------------------
__global__ void wconv_tr_bf16(const float* __restrict__ src,
                              unsigned short* __restrict__ dst,
                              int K, int N, long total) {
    long i = (long)blockIdx.x * 256 + threadIdx.x;
    if (i >= total) return;
    long kn = (long)K * N;
    long b  = i / kn;
    long r  = i - b * kn;
    int  n  = (int)(r / K);
    int  k  = (int)(r - (long)n * K);
    dst[b * kn + (long)n * K + k] = bf16_bits(src[b * kn + (long)k * N + n]);
}

// ---------------------------------------------------------------------------
// Prep B: pad/transpose third-layer weights to w3t[7][16][256] bf16.
//   heads 0..5: row r (=output col) < 3 -> bW3[n][k][r]; rows 3..15 zero
//   head 6 (speed): row 0 -> so_W3[k]; rows 1..15 zero
// ---------------------------------------------------------------------------
__global__ void w3_prep(const float* __restrict__ bW3,
                        const float* __restrict__ so_W3,
                        unsigned short* __restrict__ dst) {
    int i = blockIdx.x * 256 + threadIdx.x;          // over 7*16*256 = 28672
    if (i >= 7 * 16 * 256) return;
    int n = i / (16 * 256);
    int r = (i / 256) & 15;
    int k = i & 255;
    float v = 0.f;
    if (n < 6) { if (r < 3) v = bW3[n * 768 + k * 3 + r]; }
    else       { if (r == 0) v = so_W3[k]; }
    dst[i] = bf16_bits(v);
}

// ---------------------------------------------------------------------------
// Fused CILRS forward. 256 threads (8 waves); each wave owns 16 batch rows.
// LDS (dynamic, bf16 elems):
//   Wb[2][256][40]   weight slice, double buffered (WG-shared)
//   Hb[8][16][264]   per-wave hidden activations
//   At[8][16][648]   per-wave A-tile: concat(embedding, speed_latent) bf16
// Total 274,432 B  (<= 320 KB WGP LDS)
// ---------------------------------------------------------------------------
__global__ __launch_bounds__(256) void cilrs_fused(
    const float* __restrict__ emb,
    const float* __restrict__ speed,
    const int*   __restrict__ command,
    const float* __restrict__ si_W1, const float* __restrict__ si_b1,
    const float* __restrict__ si_b2,
    const float* __restrict__ bb1,  const float* __restrict__ bb2,
    const float* __restrict__ bb3,
    const float* __restrict__ so_b1, const float* __restrict__ so_b2,
    const float* __restrict__ so_b3,
    const unsigned short* __restrict__ wsi_W2t,
    const unsigned short* __restrict__ wso_W1t,
    const unsigned short* __restrict__ wso_W2t,
    const unsigned short* __restrict__ wbW1t,
    const unsigned short* __restrict__ wbW2t,
    const unsigned short* __restrict__ w3t,
    float* __restrict__ out)
{
    extern __shared__ unsigned short lds[];
    unsigned short* Wb = lds;                        // 2*256*40 = 20480 elems
    unsigned short* Hb = lds + 2 * 256 * 40;         // 8*16*264 = 33792 elems
    unsigned short* At = Hb + 8 * 16 * 264;          // 8*16*648 = 82944 elems

    const int tid  = threadIdx.x;
    const int wave = tid >> 5;
    const int lane = tid & 31;
    const int lh   = lane >> 4;       // half-wave selector (K split / M split)
    const int ln   = lane & 15;
    const int row0 = blockIdx.x * 128 + wave * 16;

    unsigned short* HbW = Hb + wave * (16 * 264);
    unsigned short* AtW = At + wave * (16 * 648);

    // ------------------------------------------------------------------
    // Stage embedding tile [16][512] f32 -> AtW[:, 0:512] bf16 (once).
    // ------------------------------------------------------------------
    for (int r = 0; r < 16; r++) {
        const float4* src = (const float4*)(emb + (size_t)(row0 + r) * 512 + lane * 16);
        float4 f0 = src[0], f1 = src[1], f2 = src[2], f3 = src[3];
        uint4 a = {pk2(f0.x, f0.y), pk2(f0.z, f0.w), pk2(f1.x, f1.y), pk2(f1.z, f1.w)};
        uint4 b = {pk2(f2.x, f2.y), pk2(f2.z, f2.w), pk2(f3.x, f3.y), pk2(f3.z, f3.w)};
        uint4* dst = (uint4*)(AtW + r * 648 + lane * 16);
        dst[0] = a;
        dst[1] = b;
    }

    // ------------------------------------------------------------------
    // speed_in: s = relu(speed*W1 + b1) @ W2 + b2 -> AtW[:, 512:640] bf16
    // ------------------------------------------------------------------
    {
        float spd = speed[row0 + ln];
        v8f accS[8];
        #pragma unroll
        for (int j = 0; j < 8; j++) {
            #pragma unroll
            for (int v = 0; v < 8; v++) accS[j][v] = 0.f;
        }
        for (int c = 0; c < 8; c++) {            // K = 256 -> 8 chunks of 32
            v8u hu;
            #pragma unroll
            for (int v = 0; v < 8; v++) {
                int k = c * 32 + ((v < 4) ? v * 2 : 16 + (v - 4) * 2) + lh * 8;
                float h0 = fmaxf(fmaf(spd, si_W1[k],     si_b1[k]),     0.f);
                float h1 = fmaxf(fmaf(spd, si_W1[k + 1], si_b1[k + 1]), 0.f);
                hu[v] = pk2(h0, h1);
            }
            v16bf hv = __builtin_bit_cast(v16bf, hu);
            #pragma unroll
            for (int j = 0; j < 8; j++) {        // N = 128 -> 8 col blocks
                const unsigned short* wp =
                    wsi_W2t + (size_t)(j * 16 + ln) * 256 + c * 32 + lh * 8;
                uint4 q0 = *(const uint4*)wp;
                uint4 q1 = *(const uint4*)(wp + 16);
                v8u bu = {q0.x, q0.y, q0.z, q0.w, q1.x, q1.y, q1.z, q1.w};
                v16bf bv = __builtin_bit_cast(v16bf, bu);
                accS[j] = __builtin_amdgcn_wmma_f32_16x16x32_bf16(
                    false, hv, false, bv, (short)0, accS[j], false, false);
            }
        }
        #pragma unroll
        for (int j = 0; j < 8; j++) {
            float bb = si_b2[j * 16 + ln];
            #pragma unroll
            for (int v = 0; v < 8; v++)
                AtW[(v + 8 * lh) * 648 + 512 + j * 16 + ln] =
                    bf16_bits(accS[j][v] + bb);   // no ReLU (linear layer)
        }
        asm volatile("" ::: "memory");
    }

    v8f acc[16];   // 16x256 f32 accumulator tile (128 VGPRs)

    // Cooperative weight-slice staging: Wt row-major [256][K] bf16.
    auto stage = [&](const unsigned short* Wt, int K, int kk, int buf) {
        const unsigned short* src = Wt + (size_t)tid * K + kk * 32;
        unsigned short*       dst = Wb + buf * (256 * 40) + tid * 40;
#if USE_ASYNC_LDS
        gv4i_p g = (gv4i_p)(v4i*)(void*)(const_cast<unsigned short*>(src));
        lv4i_p l = (lv4i_p)(v4i*)(void*)dst;
        __builtin_amdgcn_global_load_async_to_lds_b128(g, l,  0, 0);
        __builtin_amdgcn_global_load_async_to_lds_b128(g, l, 16, 0);
        __builtin_amdgcn_global_load_async_to_lds_b128(g, l, 32, 0);
        __builtin_amdgcn_global_load_async_to_lds_b128(g, l, 48, 0);
#else
        const uint4* s4 = (const uint4*)src;
        uint4*       d4 = (uint4*)dst;
        d4[0] = s4[0]; d4[1] = s4[1]; d4[2] = s4[2]; d4[3] = s4[3];
#endif
    };
    auto stage_wait = [&]() {
#if USE_ASYNC_LDS
        __builtin_amdgcn_s_wait_asynccnt(0);
#endif
    };

    // One 16x256 GEMM: A from LDS (base Ab, row stride Astr), W streamed
    // through the double-buffered Wb slice shared by all 8 waves.
    auto run_gemm = [&](const unsigned short* Wt, int K,
                        const unsigned short* Ab, int Astr) {
        #pragma unroll
        for (int jb = 0; jb < 16; jb++) {
            #pragma unroll
            for (int v = 0; v < 8; v++) acc[jb][v] = 0.f;
        }
        const int nk = K >> 5;
        stage(Wt, K, 0, 0);
        stage_wait();
        __syncthreads();
        for (int kk = 0; kk < nk; kk++) {
            int cur = kk & 1;
            if (kk + 1 < nk) stage(Wt, K, kk + 1, cur ^ 1);
            const unsigned short* p = Ab + ln * Astr + kk * 32 + lh * 8;
            uint4 q0 = *(const uint4*)p;
            uint4 q1 = *(const uint4*)(p + 16);
            v8u au = {q0.x, q0.y, q0.z, q0.w, q1.x, q1.y, q1.z, q1.w};
            v16bf av = __builtin_bit_cast(v16bf, au);
            #pragma unroll
            for (int jb = 0; jb < 16; jb++) {
                const unsigned short* bp =
                    Wb + cur * (256 * 40) + (jb * 16 + ln) * 40 + lh * 8;
                uint4 b0 = *(const uint4*)bp;
                uint4 b1 = *(const uint4*)(bp + 16);
                v8u bu = {b0.x, b0.y, b0.z, b0.w, b1.x, b1.y, b1.z, b1.w};
                v16bf bv = __builtin_bit_cast(v16bf, bu);
                acc[jb] = __builtin_amdgcn_wmma_f32_16x16x32_bf16(
                    false, av, false, bv, (short)0, acc[jb], false, false);
            }
            stage_wait();
            __syncthreads();
        }
    };

    auto epilogue = [&](const float* bias) {   // bias + ReLU -> HbW bf16
        #pragma unroll
        for (int jb = 0; jb < 16; jb++) {
            float bv = bias[jb * 16 + ln];
            #pragma unroll
            for (int v = 0; v < 8; v++) {
                float x = fmaxf(acc[jb][v] + bv, 0.f);
                HbW[(v + 8 * lh) * 264 + jb * 16 + ln] = bf16_bits(x);
            }
        }
        asm volatile("" ::: "memory");
    };

    // n = 0..5: command branches; n = 6: speed head
    for (int n = 0; n < 7; n++) {
        const unsigned short* W1t = (n < 6) ? (wbW1t + (size_t)n * (256 * 640)) : wso_W1t;
        const float*          b1p = (n < 6) ? (bb1 + n * 256) : so_b1;
        const unsigned short* W2t = (n < 6) ? (wbW2t + (size_t)n * (256 * 256)) : wso_W2t;
        const float*          b2p = (n < 6) ? (bb2 + n * 256) : so_b2;
        __builtin_prefetch(W1t, 0, 1);

        run_gemm(W1t, 640, AtW, 648);
        epilogue(b1p);
        run_gemm(W2t, 256, HbW, 264);
        epilogue(b2p);

        // ---- third layer as a WMMA mini-GEMM: h2[16x256] @ W3pad[256x16] ----
        {
            const unsigned short* W3t = w3t + (size_t)n * (16 * 256);
            v8f acc3;
            #pragma unroll
            for (int v = 0; v < 8; v++) acc3[v] = 0.f;
            #pragma unroll
            for (int kk = 0; kk < 8; kk++) {
                const unsigned short* p = HbW + ln * 264 + kk * 32 + lh * 8;
                uint4 q0 = *(const uint4*)p;
                uint4 q1 = *(const uint4*)(p + 16);
                v8u au = {q0.x, q0.y, q0.z, q0.w, q1.x, q1.y, q1.z, q1.w};
                v16bf av = __builtin_bit_cast(v16bf, au);
                const unsigned short* wp = W3t + (size_t)ln * 256 + kk * 32 + lh * 8;
                uint4 b0 = *(const uint4*)wp;
                uint4 b1 = *(const uint4*)(wp + 16);
                v8u bu = {b0.x, b0.y, b0.z, b0.w, b1.x, b1.y, b1.z, b1.w};
                v16bf bv = __builtin_bit_cast(v16bf, bu);
                acc3 = __builtin_amdgcn_wmma_f32_16x16x32_bf16(
                    false, av, false, bv, (short)0, acc3, false, false);
            }
            if (n < 6) {
                if (ln < 3) {
                    float b3v = bb3[n * 3 + ln];
                    #pragma unroll
                    for (int v = 0; v < 8; v++) {
                        int row = row0 + v + 8 * lh;
                        if (command[row] - 1 == n) {
                            float x = acc3[v] + b3v;
                            out[(size_t)row * 3 + ln] = 1.f / (1.f + __expf(-x));
                        }
                    }
                }
            } else {
                if (ln == 0) {
                    float b3v = so_b3[0];
                    #pragma unroll
                    for (int v = 0; v < 8; v++)
                        out[(size_t)B_ROWS * 3 + row0 + v + 8 * lh] = acc3[v] + b3v;
                }
            }
        }
    }
}

// ---------------------------------------------------------------------------
extern "C" void kernel_launch(void* const* d_in, const int* in_sizes, int n_in,
                              void* d_out, int out_size, void* d_ws, size_t ws_size,
                              hipStream_t stream) {
    const float* emb     = (const float*)d_in[0];
    const float* speed   = (const float*)d_in[1];
    const int*   command = (const int*)d_in[2];
    const float* si_W1 = (const float*)d_in[3];
    const float* si_b1 = (const float*)d_in[4];
    const float* si_W2 = (const float*)d_in[5];
    const float* si_b2 = (const float*)d_in[6];
    const float* bW1   = (const float*)d_in[7];
    const float* bb1   = (const float*)d_in[8];
    const float* bW2   = (const float*)d_in[9];
    const float* bb2   = (const float*)d_in[10];
    const float* bW3   = (const float*)d_in[11];
    const float* bb3   = (const float*)d_in[12];
    const float* so_W1 = (const float*)d_in[13];
    const float* so_b1 = (const float*)d_in[14];
    const float* so_W2 = (const float*)d_in[15];
    const float* so_b2 = (const float*)d_in[16];
    const float* so_W3 = (const float*)d_in[17];
    const float* so_b3 = (const float*)d_in[18];
    float* out = (float*)d_out;

    // workspace layout (bf16 elems): transposed weights for WMMA B-fragments
    unsigned short* ws = (unsigned short*)d_ws;
    unsigned short* wsi_W2t = ws;                 // 128*256            = 32768
    unsigned short* wso_W1t = ws + 32768;         // 256*640            = 163840
    unsigned short* wso_W2t = ws + 196608;        // 256*256            = 65536
    unsigned short* wbW1t   = ws + 262144;        // 6*256*640          = 983040
    unsigned short* wbW2t   = ws + 1245184;       // 6*256*256          = 393216
    unsigned short* ww3t    = ws + 1638400;       // 7*16*256           = 28672
    // total 1,667,072 bf16 = 3.34 MB of scratch

    auto tconv = [&](const float* s, unsigned short* d, int K, int N, int batch) {
        long total = (long)batch * K * N;
        int blocks = (int)((total + 255) / 256);
        wconv_tr_bf16<<<blocks, 256, 0, stream>>>(s, d, K, N, total);
    };
    tconv(si_W2, wsi_W2t, 256, 128, 1);
    tconv(so_W1, wso_W1t, 640, 256, 1);
    tconv(so_W2, wso_W2t, 256, 256, 1);
    tconv(bW1,   wbW1t,   640, 256, 6);
    tconv(bW2,   wbW2t,   256, 256, 6);
    w3_prep<<<(7 * 16 * 256 + 255) / 256, 256, 0, stream>>>(bW3, so_W3, ww3t);

    size_t smem = (size_t)(2 * 256 * 40 + 8 * 16 * 264 + 8 * 16 * 648)
                  * sizeof(unsigned short);   // 274,432 B
    (void)hipFuncSetAttribute((const void*)cilrs_fused,
                              hipFuncAttributeMaxDynamicSharedMemorySize, (int)smem);
    cilrs_fused<<<B_ROWS / 128, 256, smem, stream>>>(
        emb, speed, command,
        si_W1, si_b1, si_b2,
        bb1, bb2, bb3,
        so_b1, so_b2, so_b3,
        wsi_W2t, wso_W1t, wso_W2t, wbW1t, wbW2t, ww3t,
        out);
}
